// CRF_32006096290490
// MI455X (gfx1250) — matrix operand order repaired and checked by reading, not compiled, not run on hardware
//
#include <hip/hip_runtime.h>
#include <hip/hip_bf16.h>
#include <math.h>

typedef __attribute__((ext_vector_type(16))) _Float16 v16h;
typedef __attribute__((ext_vector_type(8)))  _Float16 v8h;
typedef __attribute__((ext_vector_type(8)))  float    v8f;

#define NUM_TAGS 256
#define BATCH    64
#define TLEN     512
#define ROWS     16   // batches per workgroup (= WMMA M)
#define APAD     8    // f32 row pad  -> stride 264 dwords, kills bank conflicts
#define VPAD     8    // f16 row pad  -> stride 528 bytes

// ---------------------------------------------------------------------------
// Forward algorithm (log-partition) in the exp domain:
//   alpha_j <- m + log( sum_i exp(alpha_i - m) * E_ij ) + emit_j,  E = exp(T)
// One WG = 16 batches, 16 wave32s. Wave w owns output columns [16w,16w+16)
// and keeps its 256x16 slice of E as 8 resident f16 B-fragments (64 VGPRs).
// Row r (= wave r) max is a wave-local shuffle butterfly => 2 barriers/step.
// ---------------------------------------------------------------------------
__global__ __launch_bounds__(512)
void crf_forward_kernel(const float* __restrict__ emis,
                        const float* __restrict__ trans,
                        const float* __restrict__ startT,
                        const float* __restrict__ endT,
                        const unsigned char* __restrict__ mask,
                        float* __restrict__ partition)
{
    __shared__ float         sAlpha[ROWS][NUM_TAGS + APAD];
    __shared__ _Float16      sV[ROWS][NUM_TAGS + VPAD];
    __shared__ float         sM[ROWS];
    __shared__ unsigned char sMask[ROWS];

    const int tid  = threadIdx.x;
    const int wave = tid >> 5;     // 0..15 : wave id == owned row in row phases
    const int lane = tid & 31;
    const int l16  = lane & 15;
    const int hi   = lane >> 4;
    const int b0   = blockIdx.x * ROWS;

    // ---- resident B fragments: E = exp(transitions), column tile = wave*16 ----
    // B layout (K=32 x N=16, f16): halves 0-7 -> K = hi*8 + h,
    // halves 8-15 -> K = 16 + hi*8 + (h-8); column N = lane&15.
    v16h bfrag[8];
    {
        const int col = wave * 16 + l16;
        #pragma unroll
        for (int kb = 0; kb < 8; ++kb)
            #pragma unroll
            for (int h = 0; h < 16; ++h) {
                const int k = kb * 32 + (h < 8 ? hi * 8 + h : 16 + hi * 8 + (h - 8));
                bfrag[kb][h] = (_Float16)__expf(trans[k * NUM_TAGS + col]);
            }
    }

    // ---- alpha0 = start_transitions + emissions[:,0,:] (wave writes own row) ----
    {
        const int r = wave;
        #pragma unroll
        for (int k = 0; k < 8; ++k) {
            const int c = lane * 8 + k;
            sAlpha[r][c] = startT[c] + emis[((size_t)(b0 + r) * TLEN) * NUM_TAGS + c];
        }
    }
    __syncthreads();

    for (int t = 1; t < TLEN; ++t) {
        // ---- phase A: wave-local row max (shuffle butterfly), stage V/m/mask ----
        {
            const int r = wave;
            float lm = -INFINITY;
            #pragma unroll
            for (int k = 0; k < 8; ++k)
                lm = fmaxf(lm, sAlpha[r][lane + 32 * k]);
            #pragma unroll
            for (int off = 16; off > 0; off >>= 1)
                lm = fmaxf(lm, __shfl_xor(lm, off, 32));
            if (lane == 0) {
                sM[r]    = lm;
                sMask[r] = mask[(size_t)(b0 + r) * TLEN + t];
            }
            if (t + 1 < TLEN)
                __builtin_prefetch(&emis[((size_t)(b0 + r) * TLEN + (t + 1)) * NUM_TAGS + lane * 8], 0, 0);
            #pragma unroll
            for (int k = 0; k < 8; ++k) {
                const int c = lane * 8 + k;
                sV[r][c] = (_Float16)__expf(sAlpha[r][c] - lm);
            }
        }
        __syncthreads();   // barrier 1: sV / sM / sMask visible to all waves

        // ---- phase B: U(16x16 tile) = V(16x256) x E(256x256) ----
        // Two independent accumulator chains (even/odd K-blocks) to halve the
        // dependent-WMMA depth on the step's critical path.
        // A layout: row M = lane&15; halves 0-7 -> K = kb*32 + hi*8 .. +7,
        //           halves 8-15 -> +16. Two ds_load_b128 per K-block.
        v8f acc0 = {0.f, 0.f, 0.f, 0.f, 0.f, 0.f, 0.f, 0.f};
        v8f acc1 = {0.f, 0.f, 0.f, 0.f, 0.f, 0.f, 0.f, 0.f};
        #pragma unroll
        for (int kp = 0; kp < 4; ++kp) {
            const int kbe = 2 * kp, kbo = 2 * kp + 1;
            const v8h lo0 = *(const v8h*)&sV[l16][kbe * 32 + hi * 8];
            const v8h hv0 = *(const v8h*)&sV[l16][kbe * 32 + 16 + hi * 8];
            const v16h a0 = __builtin_shufflevector(lo0, hv0,
                                0,1,2,3,4,5,6,7,8,9,10,11,12,13,14,15);
            acc0 = __builtin_amdgcn_wmma_f32_16x16x32_f16(false, a0, false, bfrag[kbe],
                                                          (short)0, acc0, false, false);
            const v8h lo1 = *(const v8h*)&sV[l16][kbo * 32 + hi * 8];
            const v8h hv1 = *(const v8h*)&sV[l16][kbo * 32 + 16 + hi * 8];
            const v16h a1 = __builtin_shufflevector(lo1, hv1,
                                0,1,2,3,4,5,6,7,8,9,10,11,12,13,14,15);
            acc1 = __builtin_amdgcn_wmma_f32_16x16x32_f16(false, a1, false, bfrag[kbo],
                                                          (short)0, acc1, false, false);
        }

        // ---- phase C: alpha_new = m + log(u) + emit ; masked update ----
        // D layout: VGPR q -> row q + 8*hi, col = wave*16 + (lane&15).
        {
            const int col = wave * 16 + l16;
            #pragma unroll
            for (int q = 0; q < 8; ++q) {
                const int r    = q + hi * 8;
                const float u  = acc0[q] + acc1[q];
                const float nw = sM[r] + __logf(u)
                               + emis[((size_t)(b0 + r) * TLEN + t) * NUM_TAGS + col];
                const float od = sAlpha[r][col];
                sAlpha[r][col] = sMask[r] ? nw : od;
            }
        }
        __syncthreads();   // barrier 2: alpha write-back complete
    }

    // ---- partition = logsumexp(alpha + end_transitions), pure shuffle reduce ----
    {
        const int r = wave;
        float lm = -INFINITY;
        #pragma unroll
        for (int k = 0; k < 8; ++k) {
            const int c = lane + 32 * k;
            lm = fmaxf(lm, sAlpha[r][c] + endT[c]);
        }
        #pragma unroll
        for (int off = 16; off > 0; off >>= 1)
            lm = fmaxf(lm, __shfl_xor(lm, off, 32));
        float ls = 0.f;
        #pragma unroll
        for (int k = 0; k < 8; ++k) {
            const int c = lane + 32 * k;
            ls += __expf(sAlpha[r][c] + endT[c] - lm);
        }
        #pragma unroll
        for (int off = 16; off > 0; off >>= 1)
            ls += __shfl_xor(ls, off, 32);
        if (lane == 0)
            partition[b0 + r] = lm + __logf(ls);
    }
}

// ---------------------------------------------------------------------------
// Gold path score: one block per batch, threads stride over time.
// ---------------------------------------------------------------------------
__global__ __launch_bounds__(256)
void crf_score_kernel(const float* __restrict__ emis,
                      const float* __restrict__ trans,
                      const float* __restrict__ startT,
                      const float* __restrict__ endT,
                      const int* __restrict__ tags,
                      const unsigned char* __restrict__ mask,
                      float* __restrict__ score)
{
    __shared__ float sSum[256];
    __shared__ int   sCnt[256];
    const int b = blockIdx.x, tid = threadIdx.x;
    float sum = 0.f;
    int   cnt = 0;
    for (int t = tid; t < TLEN; t += 256) {
        const int tg = tags[b * TLEN + t];
        const unsigned char m = mask[b * TLEN + t];
        cnt += m ? 1 : 0;
        if (t >= 1 && m) {
            const int tp = tags[b * TLEN + t - 1];
            sum += trans[tp * NUM_TAGS + tg]
                 + emis[((size_t)b * TLEN + t) * NUM_TAGS + tg];
        }
    }
    sSum[tid] = sum;
    sCnt[tid] = cnt;
    __syncthreads();
    for (int s = 128; s > 0; s >>= 1) {
        if (tid < s) { sSum[tid] += sSum[tid + s]; sCnt[tid] += sCnt[tid + s]; }
        __syncthreads();
    }
    if (tid == 0) {
        const int tg0 = tags[b * TLEN];
        float sc = sSum[0] + startT[tg0]
                 + emis[(size_t)b * TLEN * NUM_TAGS + tg0];
        const int last = sCnt[0] - 1;
        sc += endT[tags[b * TLEN + last]];
        score[b] = sc;
    }
}

// ---------------------------------------------------------------------------
// mean(partition - score)
// ---------------------------------------------------------------------------
__global__ __launch_bounds__(64)
void crf_finalize(const float* __restrict__ partition,
                  const float* __restrict__ score,
                  float* __restrict__ out)
{
    __shared__ float s[64];
    const int tid = threadIdx.x;
    s[tid] = partition[tid] - score[tid];
    __syncthreads();
    for (int k = 32; k > 0; k >>= 1) {
        if (tid < k) s[tid] += s[tid + k];
        __syncthreads();
    }
    if (tid == 0) out[0] = s[0] / (float)BATCH;
}

extern "C" void kernel_launch(void* const* d_in, const int* in_sizes, int n_in,
                              void* d_out, int out_size, void* d_ws, size_t ws_size,
                              hipStream_t stream) {
    (void)in_sizes; (void)n_in; (void)out_size; (void)ws_size;
    const float*         emis   = (const float*)d_in[0];         // [64,512,256]
    const float*         trans  = (const float*)d_in[1];         // [256,256]
    const float*         startT = (const float*)d_in[2];         // [256]
    const float*         endT   = (const float*)d_in[3];         // [256]
    const int*           tags   = (const int*)d_in[4];           // [64,512]
    const unsigned char* mask   = (const unsigned char*)d_in[5]; // [64,512] bool

    float* wsf       = (float*)d_ws;
    float* partition = wsf;        // 64 floats
    float* score     = wsf + 64;   // 64 floats

    crf_forward_kernel<<<BATCH / ROWS, 512, 0, stream>>>(emis, trans, startT, endT,
                                                         mask, partition);
    crf_score_kernel<<<BATCH, 256, 0, stream>>>(emis, trans, startT, endT,
                                                tags, mask, score);
    crf_finalize<<<1, 64, 0, stream>>>(partition, score, (float*)d_out);
}
